// GRUController_7164005450349
// MI455X (gfx1250) — matrix operand order repaired and checked by reading, not compiled, not run on hardware
//
#include <hip/hip_runtime.h>

// ---------------------------------------------------------------------------
// Persistent fused GRU encoder/decoder for MI455X (gfx1250, wave32).
//  - 1 wave = 16 batch rows, full recurrence kept on-chip.
//  - All per-step GEMMs via v_wmma_f32_16x16x32_f16 (A = state frags in VGPRs,
//    B = f16 weight fragments pre-packed in LDS).
//  - D-layout -> A-layout state conversion through per-wave LDS scratch.
//  - Activations use native v_exp_f32 / v_rcp_f32 / v_tanh_f32 (no IEEE div).
// ---------------------------------------------------------------------------

typedef __attribute__((ext_vector_type(16))) _Float16 v16h;
typedef __attribute__((ext_vector_type(8)))  _Float16 v8h;
typedef __attribute__((ext_vector_type(8)))  float    v8f;

#define HID      64
#define G3       192
#define B_TOT    8192
#define T_IN     512
#define T_OUT    180
#define WAVES    8
#define THREADS  (WAVES * 32)
#define NBLOCKS  (B_TOT / (16 * WAVES))   // 64

// ---- LDS layout (bytes) ----
// weight fragments: 6 matrices (192x64) x 12 ntiles x 2 kchunks x (32 lanes*32B)
constexpr int SZ_WF     = 6 * 12 * 2 * 1024;            // 147456
constexpr int OFF_WF    = 0;
constexpr int OFF_EWC   = OFF_WF + SZ_WF;               // enc Wih0 compact: 12*16*4
constexpr int OFF_DWC   = OFF_EWC + 768;                // dec Wih0 compact
constexpr int OFF_BIAS  = OFF_DWC + 768;                // 4 layers * 256 floats
constexpr int OFF_OW    = OFF_BIAS + 4096;              // out_W (64) + out_b (1), padded
constexpr int OFF_H     = 153600;                       // per-wave h scratch: 8 * 2 * 2KB
constexpr int OFF_P     = OFF_H + WAVES * 4096;         // per-wave prev scratch: 8 * 16 floats
constexpr int LDS_TOTAL = OFF_P + WAVES * 64;           // ~183 KB

static __device__ __forceinline__ v8f wmma_f16(v16h a, v16h b, v8f c) {
  return __builtin_amdgcn_wmma_f32_16x16x32_f16(false, a, false, b, (short)0, c,
                                                false, false);
}

static __device__ __forceinline__ v16h zero16() {
  v16h r;
#pragma unroll
  for (int i = 0; i < 16; ++i) r[i] = (_Float16)0.0f;
  return r;
}

static __device__ __forceinline__ v8f splat8(float v) {
  v8f r;
#pragma unroll
  for (int i = 0; i < 8; ++i) r[i] = v;
  return r;
}

// sigmoid via native v_exp_f32 + v_rcp_f32 (no IEEE-div fixup sequence)
static __device__ __forceinline__ float sigf(float v) {
  return __builtin_amdgcn_rcpf(1.0f + __expf(-v));
}
// tanh: native v_tanh_f32 on gfx1250 when the builtin exists
static __device__ __forceinline__ float tanh_fast(float v) {
#if __has_builtin(__builtin_amdgcn_tanh_f32)
  return __builtin_amdgcn_tanh_f32(v);
#elif __has_builtin(__builtin_amdgcn_tanhf)
  return __builtin_amdgcn_tanhf(v);
#else
  float e = __expf(2.0f * v);
  return 1.0f - 2.0f * __builtin_amdgcn_rcpf(e + 1.0f);
#endif
}

static __device__ __forceinline__ unsigned short h2u(_Float16 h) {
  union { _Float16 h; unsigned short u; } cv; cv.h = h; return cv.u;
}

// B-fragment (32x16 f16) of a 192x64 weight matrix: tile tt (N), chunk c (K).
// LDS packing: frag base + lane*32 bytes -> two contiguous ds_load_b128.
static __device__ __forceinline__ v16h ldsB(const char* mat, int tt, int c, int lane) {
  const char* p = mat + ((tt * 2 + c) * 32 + lane) * 32;
  v16h r;
  ((v8h*)&r)[0] = *(const v8h*)(p);
  ((v8h*)&r)[1] = *(const v8h*)(p + 16);
  return r;
}

// Compact layer-0 input-weight fragment: only K=0,1 nonzero (lanes < 16).
static __device__ __forceinline__ v16h cfrag(const unsigned int* c, int tt, int lane) {
  v16h b = zero16();
  if (lane < 16) ((unsigned int*)&b)[0] = c[tt * 16 + (lane & 15)];
  return b;
}

// Convert h (D-layout, 4x v8f fp32) -> LDS f16 row-major -> A fragments (K=0..31, 32..63).
static __device__ __forceinline__ void h_to_frags(const v8f* hd, char* buf, int lane,
                                                  v16h& a0, v16h& a1) {
  const int mhi = (lane >> 4) & 1;
  const int nlo = lane & 15;
#pragma unroll
  for (int t = 0; t < 4; ++t)
#pragma unroll
    for (int j = 0; j < 8; ++j) {
      int m = j + 8 * mhi;
      int n = 16 * t + nlo;
      *(_Float16*)(buf + m * 128 + n * 2) = (_Float16)hd[t][j];
    }
  asm volatile("s_wait_dscnt 0" ::: "memory");
  const char* p = buf + nlo * 128 + mhi * 32;
  ((v8h*)&a0)[0] = *(const v8h*)(p);
  ((v8h*)&a0)[1] = *(const v8h*)(p + 16);
  ((v8h*)&a1)[0] = *(const v8h*)(p + 64);
  ((v8h*)&a1)[1] = *(const v8h*)(p + 80);
  asm volatile("s_wait_dscnt 0" ::: "memory");
}

// One GRU layer for a 16-row batch tile.
// COMPACT: input matmul uses zero-padded-K compact fragment (layer 0).
template <bool COMPACT>
static __device__ __forceinline__ void gru_layer(v8f* hd, v16h ia0, v16h ia1,
                                                 const char* wiFrag,
                                                 const unsigned int* wiC,
                                                 v16h ha0, v16h ha1,
                                                 const char* whFrag,
                                                 const float* bias, int lane) {
  const int nlo = lane & 15;
#pragma unroll
  for (int t4 = 0; t4 < 4; ++t4) {
    v8f aR = splat8(bias[16 * t4 + nlo]);        // bih_r + bhh_r
    v8f aZ = splat8(bias[64 + 16 * t4 + nlo]);   // bih_z + bhh_z
    v8f aI = splat8(bias[128 + 16 * t4 + nlo]);  // bih_n
    v8f aN = splat8(bias[192 + 16 * t4 + nlo]);  // bhh_n
    if (COMPACT) {
      aR = wmma_f16(ia0, cfrag(wiC, t4, lane), aR);
      aZ = wmma_f16(ia0, cfrag(wiC, 4 + t4, lane), aZ);
      aI = wmma_f16(ia0, cfrag(wiC, 8 + t4, lane), aI);
    } else {
      aR = wmma_f16(ia0, ldsB(wiFrag, t4, 0, lane), aR);
      aR = wmma_f16(ia1, ldsB(wiFrag, t4, 1, lane), aR);
      aZ = wmma_f16(ia0, ldsB(wiFrag, 4 + t4, 0, lane), aZ);
      aZ = wmma_f16(ia1, ldsB(wiFrag, 4 + t4, 1, lane), aZ);
      aI = wmma_f16(ia0, ldsB(wiFrag, 8 + t4, 0, lane), aI);
      aI = wmma_f16(ia1, ldsB(wiFrag, 8 + t4, 1, lane), aI);
    }
    aR = wmma_f16(ha0, ldsB(whFrag, t4, 0, lane), aR);
    aR = wmma_f16(ha1, ldsB(whFrag, t4, 1, lane), aR);
    aZ = wmma_f16(ha0, ldsB(whFrag, 4 + t4, 0, lane), aZ);
    aZ = wmma_f16(ha1, ldsB(whFrag, 4 + t4, 1, lane), aZ);
    aN = wmma_f16(ha0, ldsB(whFrag, 8 + t4, 0, lane), aN);
    aN = wmma_f16(ha1, ldsB(whFrag, 8 + t4, 1, lane), aN);
#pragma unroll
    for (int j = 0; j < 8; ++j) {
      float r = sigf(aR[j]);
      float z = sigf(aZ[j]);
      float n = tanh_fast(aI[j] + r * aN[j]);
      hd[t4][j] = (1.0f - z) * n + z * hd[t4][j];
    }
  }
}

__global__ void __launch_bounds__(THREADS, 1)
gru_seq2seq_kernel(const float* __restrict__ x,
                   const float* eWih0, const float* eWhh0,
                   const float* ebih0, const float* ebhh0,
                   const float* eWih1, const float* eWhh1,
                   const float* ebih1, const float* ebhh1,
                   const float* dWih0, const float* dWhh0,
                   const float* dbih0, const float* dbhh0,
                   const float* dWih1, const float* dWhh1,
                   const float* dbih1, const float* dbhh1,
                   const float* outW, const float* outB,
                   float* __restrict__ out) {
  extern __shared__ char smem[];
  const int tid  = threadIdx.x;
  const int lane = tid & 31;
  const int wave = tid >> 5;
  const int nlo  = lane & 15;

  // ------------------- build LDS-resident weights -------------------------
  {
    _Float16* wf = (_Float16*)(smem + OFF_WF);
    const float* mats[6] = {eWhh0, eWih1, eWhh1, dWhh0, dWih1, dWhh1};
    for (int i = tid; i < 6 * 12288; i += THREADS) {
      int mi = i / 12288;
      int r  = i % 12288;           // = f*512 + l*16 + e
      int f = r >> 9, l = (r >> 4) & 31, e = r & 15;
      int t = f >> 1, c = f & 1;
      int j = 16 * t + (l & 15);
      int k = 32 * c + 16 * (l >> 4) + e;
      wf[i] = (_Float16)mats[mi][j * 64 + k];
    }
    unsigned int* ewc = (unsigned int*)(smem + OFF_EWC);
    unsigned int* dwc = (unsigned int*)(smem + OFF_DWC);
    for (int i = tid; i < 192; i += THREADS) {
      int t = i >> 4, n = i & 15;
      int j = 16 * t + n;
      unsigned int lo = h2u((_Float16)eWih0[j * 2 + 0]);
      unsigned int hi = h2u((_Float16)eWih0[j * 2 + 1]);
      ewc[i] = lo | (hi << 16);
      dwc[i] = (unsigned int)h2u((_Float16)dWih0[j]);   // K=1, elem1 = 0
    }
    float* bb = (float*)(smem + OFF_BIAS);
    const float* bihs[4] = {ebih0, ebih1, dbih0, dbih1};
    const float* bhhs[4] = {ebhh0, ebhh1, dbhh0, dbhh1};
    for (int i = tid; i < 4 * 256; i += THREADS) {
      int L = i >> 8, r = i & 255;
      float v;
      if (r < 128)      v = bihs[L][r] + bhhs[L][r];    // fused r/z bias
      else if (r < 192) v = bihs[L][r];                 // bih_n
      else              v = bhhs[L][r - 64];            // bhh_n
      bb[i] = v;
    }
    float* ow = (float*)(smem + OFF_OW);
    for (int i = tid; i < 64; i += THREADS) ow[i] = outW[i];
    if (tid == 0) ow[64] = outB[0];
  }
  __syncthreads();

  const char* fEWhh0 = smem + OFF_WF + 0 * 24576;
  const char* fEWih1 = smem + OFF_WF + 1 * 24576;
  const char* fEWhh1 = smem + OFF_WF + 2 * 24576;
  const char* fDWhh0 = smem + OFF_WF + 3 * 24576;
  const char* fDWih1 = smem + OFF_WF + 4 * 24576;
  const char* fDWhh1 = smem + OFF_WF + 5 * 24576;
  const unsigned int* ewc = (const unsigned int*)(smem + OFF_EWC);
  const unsigned int* dwc = (const unsigned int*)(smem + OFF_DWC);
  const float* bE0 = (const float*)(smem + OFF_BIAS) + 0 * 256;
  const float* bE1 = (const float*)(smem + OFF_BIAS) + 1 * 256;
  const float* bD0 = (const float*)(smem + OFF_BIAS) + 2 * 256;
  const float* bD1 = (const float*)(smem + OFF_BIAS) + 3 * 256;
  const float* owp = (const float*)(smem + OFF_OW);

  char*  hbuf0 = smem + OFF_H + wave * 4096;
  char*  hbuf1 = hbuf0 + 2048;
  float* pbuf  = (float*)(smem + OFF_P) + wave * 16;

  const int b0 = (blockIdx.x * WAVES + wave) * 16;

  // Per-wave recurrent state.
  v8f h0d[4], h1d[4];
#pragma unroll
  for (int t = 0; t < 4; ++t) { h0d[t] = splat8(0.0f); h1d[t] = splat8(0.0f); }
  v16h h0a0 = zero16(), h0a1 = zero16();
  v16h h1a0 = zero16(), h1a1 = zero16();

  // -------------------------- encoder: 512 steps --------------------------
#pragma unroll 1
  for (int t = 0; t < T_IN; ++t) {
    v16h xa = zero16();
    if (lane < 16) {
      const float* px = x + ((size_t)(b0 + lane) * T_IN + t) * 2;
      float2 xv = *(const float2*)px;
      xa[0] = (_Float16)xv.x;
      xa[1] = (_Float16)xv.y;
      if (t + 8 < T_IN) __builtin_prefetch(px + 16, 0, 3);  // global_prefetch
    }
    gru_layer<true >(h0d, xa, xa, nullptr, ewc, h0a0, h0a1, fEWhh0, bE0, lane);
    h_to_frags(h0d, hbuf0, lane, h0a0, h0a1);
    gru_layer<false>(h1d, h0a0, h0a1, fEWih1, nullptr, h1a0, h1a1, fEWhh1, bE1, lane);
    h_to_frags(h1d, hbuf1, lane, h1a0, h1a1);
  }

  // -------------------------- decoder: 180 steps --------------------------
  float owv[4];
#pragma unroll
  for (int t = 0; t < 4; ++t) owv[t] = owp[16 * t + nlo];
  const float owb = owp[64];

  v16h pa = zero16();  // prev prediction = 0
#pragma unroll 1
  for (int s = 0; s < T_OUT; ++s) {
    gru_layer<true >(h0d, pa, pa, nullptr, dwc, h0a0, h0a1, fDWhh0, bD0, lane);
    h_to_frags(h0d, hbuf0, lane, h0a0, h0a1);
    gru_layer<false>(h1d, h0a0, h0a1, fDWih1, nullptr, h1a0, h1a1, fDWhh1, bD1, lane);
    h_to_frags(h1d, hbuf1, lane, h1a0, h1a1);

    // cv = h1 @ out_W.T + out_b : per-row dot + butterfly over 16-lane N group
    float ss[8];
#pragma unroll
    for (int j = 0; j < 8; ++j) {
      float a = 0.0f;
#pragma unroll
      for (int t4 = 0; t4 < 4; ++t4) a += h1d[t4][j] * owv[t4];
      ss[j] = a;
    }
#pragma unroll
    for (int mask = 1; mask < 16; mask <<= 1)
#pragma unroll
      for (int j = 0; j < 8; ++j) ss[j] += __shfl_xor(ss[j], mask, 32);

    if ((lane & 15) == 0) {
#pragma unroll
      for (int j = 0; j < 8; ++j) {
        float cv = ss[j] + owb;
        int m = j + 8 * (lane >> 4);
        out[(size_t)(b0 + m) * T_OUT + s] = cv;
        pbuf[m] = cv;
      }
    }
    asm volatile("s_wait_dscnt 0" ::: "memory");
    pa = zero16();
    if (lane < 16) pa[0] = (_Float16)pbuf[lane];
  }
}

extern "C" void kernel_launch(void* const* d_in, const int* in_sizes, int n_in,
                              void* d_out, int out_size, void* d_ws, size_t ws_size,
                              hipStream_t stream) {
  (void)in_sizes; (void)n_in; (void)out_size; (void)d_ws; (void)ws_size;
  const float* p[19];
  for (int i = 0; i < 19; ++i) p[i] = (const float*)d_in[i];

  hipFuncSetAttribute((const void*)gru_seq2seq_kernel,
                      hipFuncAttributeMaxDynamicSharedMemorySize, LDS_TOTAL);

  hipLaunchKernelGGL(gru_seq2seq_kernel, dim3(NBLOCKS), dim3(THREADS), LDS_TOTAL,
                     stream,
                     p[0],  p[1],  p[2],  p[3],  p[4],  p[5],  p[6],  p[7],
                     p[8],  p[9],  p[10], p[11], p[12], p[13], p[14], p[15],
                     p[16], p[17], p[18], (float*)d_out);
}
